// MolecularEncoder_78855599555022
// MI455X (gfx1250) — compile-verified, matrix operand order checked by reading, not run on hardware
//
#include <hip/hip_runtime.h>

#define H 128
#define DIN 6
#define NOUT 256
#define NLAYERS 4

typedef __attribute__((ext_vector_type(16))) _Float16 v16h;
typedef __attribute__((ext_vector_type(8)))  _Float16 v8h;
typedef __attribute__((ext_vector_type(4)))  _Float16 v4h;
typedef __attribute__((ext_vector_type(8)))  float    v8f;
typedef __attribute__((ext_vector_type(4)))  float    v4f;

union AB { v16h f; v8h h[2]; };

// chunk group for K index: 16-byte chunks ordered to match WMMA A/B lane layout
__device__ __host__ __forceinline__ int kgrp(int k) {
    return (k >> 5) * 4 + ((k >> 3) & 3);
}

// ---------------------------------------------------------------------------
// Pack a f32 weight W[K][Ncols] into f16 "fragment-chunk" order:
// dst[(kgrp(k)*Ncols + n)*8 + (k&7)] = (f16) W[k][n]
// A B-fragment load then becomes two contiguous 16B reads per lane.
// ---------------------------------------------------------------------------
__global__ void k_pack_w(const float* __restrict__ src, _Float16* __restrict__ dst,
                         int K, int Ncols) {
    int i = blockIdx.x * blockDim.x + threadIdx.x;
    if (i >= K * Ncols) return;
    int k = i / Ncols;
    int n = i - k * Ncols;
    dst[((size_t)kgrp(k) * Ncols + n) * 8 + (k & 7)] = (_Float16)src[i];
}

// ---------------------------------------------------------------------------
// Node encoder: h = x @ W_enc + b_enc ; also seeds agg = h (GIN eps=0 term).
// One wave per node; lane owns 4 of the 128 features; W_enc column slice is
// loop-invariant and lives in registers.
// ---------------------------------------------------------------------------
__global__ void __launch_bounds__(256)
k_encoder(const float* __restrict__ x, const float* __restrict__ Wenc,
          const float* __restrict__ benc, float* __restrict__ h,
          float* __restrict__ agg, int nNodes) {
    int gtid = blockIdx.x * blockDim.x + threadIdx.x;
    int lane = gtid & 31;
    int wid  = gtid >> 5;
    int nw   = (gridDim.x * blockDim.x) >> 5;
    int c0   = lane * 4;

    v4f w[DIN];
#pragma unroll
    for (int d = 0; d < DIN; ++d) w[d] = *(const v4f*)(Wenc + d * H + c0);
    v4f b = *(const v4f*)(benc + c0);

    for (int n = wid; n < nNodes; n += nw) {
        const float* xr = x + (size_t)n * DIN;
        v4f acc = b;
#pragma unroll
        for (int d = 0; d < DIN; ++d) acc += xr[d] * w[d];
        *(v4f*)(h   + (size_t)n * H + c0) = acc;
        *(v4f*)(agg + (size_t)n * H + c0) = acc;
    }
}

// ---------------------------------------------------------------------------
// Edge scatter-add: agg[dst] += h[src]. One wave per edge, lane owns 4 floats.
// f32 hardware atomics (global_atomic_add_f32) at agent scope.
// ---------------------------------------------------------------------------
__global__ void __launch_bounds__(256)
k_scatter(const float* __restrict__ h, const int* __restrict__ srcIdx,
          const int* __restrict__ dstIdx, float* __restrict__ agg, int nEdges) {
    int gtid = blockIdx.x * blockDim.x + threadIdx.x;
    int lane = gtid & 31;
    int wid  = gtid >> 5;
    int nw   = (gridDim.x * blockDim.x) >> 5;
    int c0   = lane * 4;

    for (int e = wid; e < nEdges; e += nw) {
        int s = srcIdx[e];
        int d = dstIdx[e];
        v4f v = *(const v4f*)(h + (size_t)s * H + c0);
        float* p = agg + (size_t)d * H + c0;
        __hip_atomic_fetch_add(p + 0, v.x, __ATOMIC_RELAXED, __HIP_MEMORY_SCOPE_AGENT);
        __hip_atomic_fetch_add(p + 1, v.y, __ATOMIC_RELAXED, __HIP_MEMORY_SCOPE_AGENT);
        __hip_atomic_fetch_add(p + 2, v.z, __ATOMIC_RELAXED, __HIP_MEMORY_SCOPE_AGENT);
        __hip_atomic_fetch_add(p + 3, v.w, __ATOMIC_RELAXED, __HIP_MEMORY_SCOPE_AGENT);
    }
}

// ---------------------------------------------------------------------------
// Fused GIN MLP: out = relu( relu(z@W1 + b1) @ W2 + b2 ) for a 64-row tile.
// z (f32) staged to LDS as f16 in fragment-chunk order; W1,W2 (pre-packed f16)
// staged to LDS; 8 waves, each computes one 16-row M-tile x four 16-col
// N-tiles via v_wmma_f32_16x16x32_f16. Intermediate reuses the A-tile LDS.
// Writes result to hOut and seeds aggOut (= next layer's (1+eps)*h term).
// ---------------------------------------------------------------------------
__global__ void __launch_bounds__(256)
k_gin_mlp(const float* __restrict__ agg, const _Float16* __restrict__ w1,
          const _Float16* __restrict__ w2, const float* __restrict__ b1,
          const float* __restrict__ b2, float* __restrict__ hOut,
          float* __restrict__ aggOut, int nNodes) {
    __shared__ _Float16 sA [16 * 64 * 8];   // 16 KB: A tile, then intermediate
    __shared__ _Float16 sW1[16 * H  * 8];   // 32 KB
    __shared__ _Float16 sW2[16 * H  * 8];   // 32 KB

    const int t        = threadIdx.x;
    const int lane     = t & 31;
    const int wave     = t >> 5;
    const int half     = lane >> 4;
    const int r        = lane & 15;
    const int blockRow = blockIdx.x * 64;
    const int mt       = wave >> 1;          // M tile (0..3)
    const int nbase    = (wave & 1) * 4;     // first N tile (0 or 4)

    // Stage both weight matrices (already in fragment-chunk order): 2048 v8h each
#pragma unroll
    for (int i = 0; i < 8; ++i) {
        int v = t + i * 256;
        *(v8h*)(sW1 + v * 8) = *(const v8h*)(w1 + v * 8);
        *(v8h*)(sW2 + v * 8) = *(const v8h*)(w2 + v * 8);
    }
    // Stage A tile: 64x128 f32 -> f16 chunk order (8 x v4f per thread)
#pragma unroll
    for (int i = 0; i < 8; ++i) {
        int v    = t + i * 256;              // v4f index
        int row  = v >> 5;                   // (v*4)/128
        int col  = (v & 31) * 4;
        int grow = blockRow + row;
        v4f d = {0.f, 0.f, 0.f, 0.f};
        if (grow < nNodes) d = *(const v4f*)(agg + (size_t)grow * H + col);
        v4h hd = __builtin_convertvector(d, v4h);
        *(v4h*)(sA + ((size_t)kgrp(col) * 64 + row) * 8 + (col & 7)) = hd;
    }
    __syncthreads();

    const int arow = mt * 16 + r;

    // ---- GEMM1: z @ W1 ----
    v8f acc[4] = {};
#pragma unroll
    for (int ks = 0; ks < 4; ++ks) {
        AB a;
        a.h[0] = *(const v8h*)(sA + (((ks * 4 + half)     * 64) + arow) * 8);
        a.h[1] = *(const v8h*)(sA + (((ks * 4 + 2 + half) * 64) + arow) * 8);
#pragma unroll
        for (int j = 0; j < 4; ++j) {
            int n = (nbase + j) * 16 + r;
            AB b;
            b.h[0] = *(const v8h*)(sW1 + (((ks * 4 + half)     * H) + n) * 8);
            b.h[1] = *(const v8h*)(sW1 + (((ks * 4 + 2 + half) * H) + n) * 8);
            acc[j] = __builtin_amdgcn_wmma_f32_16x16x32_f16(
                false, a.f, false, b.f, (short)0, acc[j], false, false);
        }
    }
    __syncthreads();   // all reads of sA done; safe to overwrite with intermediate

    // bias + relu -> intermediate (f16, chunk order) into sA
#pragma unroll
    for (int j = 0; j < 4; ++j) {
        int colt = (nbase + j) * 16 + r;
        float bias = b1[colt];
        int base = ((size_t)kgrp(colt) * 64) * 8 + (colt & 7);
#pragma unroll
        for (int g = 0; g < 8; ++g) {
            int row = mt * 16 + g + half * 8;
            float z = acc[j][g] + bias;
            z = z > 0.f ? z : 0.f;
            sA[base + row * 8] = (_Float16)z;
        }
    }
    __syncthreads();

    // ---- GEMM2: relu(.) @ W2 ----
    v8f acc2[4] = {};
#pragma unroll
    for (int ks = 0; ks < 4; ++ks) {
        AB a;
        a.h[0] = *(const v8h*)(sA + (((ks * 4 + half)     * 64) + arow) * 8);
        a.h[1] = *(const v8h*)(sA + (((ks * 4 + 2 + half) * 64) + arow) * 8);
#pragma unroll
        for (int j = 0; j < 4; ++j) {
            int n = (nbase + j) * 16 + r;
            AB b;
            b.h[0] = *(const v8h*)(sW2 + (((ks * 4 + half)     * H) + n) * 8);
            b.h[1] = *(const v8h*)(sW2 + (((ks * 4 + 2 + half) * H) + n) * 8);
            acc2[j] = __builtin_amdgcn_wmma_f32_16x16x32_f16(
                false, a.f, false, b.f, (short)0, acc2[j], false, false);
        }
    }

    // bias + outer relu; write new h and seed next layer's agg
#pragma unroll
    for (int j = 0; j < 4; ++j) {
        int colt = (nbase + j) * 16 + r;
        float bias = b2[colt];
#pragma unroll
        for (int g = 0; g < 8; ++g) {
            int row  = mt * 16 + g + half * 8;
            int grow = blockRow + row;
            if (grow < nNodes) {
                float z = acc2[j][g] + bias;
                z = z > 0.f ? z : 0.f;
                size_t o = (size_t)grow * H + colt;
                hOut[o]   = z;
                aggOut[o] = z;
            }
        }
    }
}

// ---------------------------------------------------------------------------
// Global mean pool. batch[] is sorted, so one wave per graph binary-searches
// its node range and reduces with zero atomics; emits f16 for the final GEMM.
// ---------------------------------------------------------------------------
__global__ void __launch_bounds__(256)
k_pool(const float* __restrict__ h, const int* __restrict__ batch,
       _Float16* __restrict__ pooled, int nNodes, int nGraphs) {
    int gtid = blockIdx.x * blockDim.x + threadIdx.x;
    int g    = gtid >> 5;
    int lane = gtid & 31;
    if (g >= nGraphs) return;

    int lo = 0, hi = nNodes;
    while (lo < hi) { int mid = (lo + hi) >> 1; if (batch[mid] < g) lo = mid + 1; else hi = mid; }
    int start = lo;
    hi = nNodes;
    while (lo < hi) { int mid = (lo + hi) >> 1; if (batch[mid] <= g) lo = mid + 1; else hi = mid; }
    int end = lo;

    int c0 = lane * 4;
    v4f acc = {0.f, 0.f, 0.f, 0.f};
    for (int i = start; i < end; ++i)
        acc += *(const v4f*)(h + (size_t)i * H + c0);

    int cnt = end - start;
    if (cnt < 1) cnt = 1;
    float inv = 1.f / (float)cnt;
    acc *= inv;
    *(v4h*)(pooled + (size_t)g * H + c0) = __builtin_convertvector(acc, v4h);
}

// ---------------------------------------------------------------------------
// Output projection: out = pooled @ W_out + b_out  ([G x 128] @ [128 x 256]).
// Same WMMA machinery; 8 waves x 8 N-tiles each, 4 K-steps.
// ---------------------------------------------------------------------------
__global__ void __launch_bounds__(256)
k_out_gemm(const _Float16* __restrict__ pooled, const _Float16* __restrict__ wo,
           const float* __restrict__ bo, float* __restrict__ out, int nGraphs) {
    __shared__ _Float16 sA[16 * 64   * 8];   // 16 KB
    __shared__ _Float16 sW[16 * NOUT * 8];   // 64 KB

    const int t        = threadIdx.x;
    const int lane     = t & 31;
    const int wave     = t >> 5;
    const int half     = lane >> 4;
    const int r        = lane & 15;
    const int blockRow = blockIdx.x * 64;
    const int mt       = wave >> 1;
    const int nbase    = (wave & 1) * 8;

    // stage pooled rows (already f16, row-major) into chunk order
#pragma unroll
    for (int i = 0; i < 4; ++i) {
        int v    = t + i * 256;            // v8h index
        int row  = v >> 4;
        int col  = (v & 15) * 8;
        int grow = blockRow + row;
        v8h d = {};
        if (grow < nGraphs) d = *(const v8h*)(pooled + (size_t)grow * H + col);
        *(v8h*)(sA + ((size_t)kgrp(col) * 64 + row) * 8) = d;
    }
    // stage packed W_out: 4096 v8h
#pragma unroll
    for (int i = 0; i < 16; ++i) {
        int v = t + i * 256;
        *(v8h*)(sW + v * 8) = *(const v8h*)(wo + v * 8);
    }
    __syncthreads();

    const int arow = mt * 16 + r;
    v8f acc[8] = {};
#pragma unroll
    for (int ks = 0; ks < 4; ++ks) {
        AB a;
        a.h[0] = *(const v8h*)(sA + (((ks * 4 + half)     * 64) + arow) * 8);
        a.h[1] = *(const v8h*)(sA + (((ks * 4 + 2 + half) * 64) + arow) * 8);
#pragma unroll
        for (int j = 0; j < 8; ++j) {
            int n = (nbase + j) * 16 + r;
            AB b;
            b.h[0] = *(const v8h*)(sW + (((ks * 4 + half)     * NOUT) + n) * 8);
            b.h[1] = *(const v8h*)(sW + (((ks * 4 + 2 + half) * NOUT) + n) * 8);
            acc[j] = __builtin_amdgcn_wmma_f32_16x16x32_f16(
                false, a.f, false, b.f, (short)0, acc[j], false, false);
        }
    }

#pragma unroll
    for (int j = 0; j < 8; ++j) {
        int colt = (nbase + j) * 16 + r;
        float bias = bo[colt];
#pragma unroll
        for (int g = 0; g < 8; ++g) {
            int grow = blockRow + mt * 16 + g + half * 8;
            if (grow < nGraphs)
                out[(size_t)grow * NOUT + colt] = acc[j][g] + bias;
        }
    }
}

// ---------------------------------------------------------------------------
extern "C" void kernel_launch(void* const* d_in, const int* in_sizes, int n_in,
                              void* d_out, int out_size, void* d_ws, size_t ws_size,
                              hipStream_t stream) {
    (void)n_in; (void)ws_size;
    const float* x     = (const float*)d_in[0];
    const int*   eidx  = (const int*)  d_in[1];
    const int*   batch = (const int*)  d_in[2];
    const float* Wenc  = (const float*)d_in[3];
    const float* benc  = (const float*)d_in[4];
    const float* W1    = (const float*)d_in[5];
    const float* b1    = (const float*)d_in[6];
    const float* W2    = (const float*)d_in[7];
    const float* b2    = (const float*)d_in[8];
    const float* Wout  = (const float*)d_in[9];
    const float* bout  = (const float*)d_in[10];
    float* out = (float*)d_out;

    const int N = in_sizes[0] / DIN;
    const int E = in_sizes[1] / 2;
    const int G = out_size / NOUT;
    const int* srcIdx = eidx;
    const int* dstIdx = eidx + E;

    char* ws = (char*)d_ws;
    size_t off = 0;
    auto carve = [&](size_t bytes) -> char* {
        char* p = ws + off;
        off = (off + bytes + 255) & ~(size_t)255;
        return p;
    };
    float*     h      = (float*)    carve((size_t)N * H * sizeof(float));
    float*     agg    = (float*)    carve((size_t)N * H * sizeof(float));
    _Float16*  w1p    = (_Float16*) carve((size_t)NLAYERS * H * H * sizeof(_Float16));
    _Float16*  w2p    = (_Float16*) carve((size_t)NLAYERS * H * H * sizeof(_Float16));
    _Float16*  wop    = (_Float16*) carve((size_t)H * NOUT * sizeof(_Float16));
    _Float16*  pooled = (_Float16*) carve((size_t)G * H * sizeof(_Float16));

    // Pack all weights to f16 fragment-chunk order (tiny, cached in L2 forever)
    int wB = (H * H + 255) / 256;
    for (int l = 0; l < NLAYERS; ++l) {
        k_pack_w<<<wB, 256, 0, stream>>>(W1 + (size_t)l * H * H, w1p + (size_t)l * H * H, H, H);
        k_pack_w<<<wB, 256, 0, stream>>>(W2 + (size_t)l * H * H, w2p + (size_t)l * H * H, H, H);
    }
    k_pack_w<<<(H * NOUT + 255) / 256, 256, 0, stream>>>(Wout, wop, H, NOUT);

    // Encoder seeds h and agg
    k_encoder<<<(N + 7) / 8, 256, 0, stream>>>(x, Wenc, benc, h, agg, N);

    const int mlpBlocks = (N + 63) / 64;
    for (int l = 0; l < NLAYERS; ++l) {
        k_scatter<<<(E + 7) / 8, 256, 0, stream>>>(h, srcIdx, dstIdx, agg, E);
        k_gin_mlp<<<mlpBlocks, 256, 0, stream>>>(agg,
                                                 w1p + (size_t)l * H * H,
                                                 w2p + (size_t)l * H * H,
                                                 b1 + l * H, b2 + l * H,
                                                 h, agg, N);
    }

    k_pool<<<(G + 7) / 8, 256, 0, stream>>>(h, batch, pooled, N, G);
    k_out_gemm<<<(G + 63) / 64, 256, 0, stream>>>(pooled, wop, bout, out, G);
}
